// DInPBlock_38525856645313
// MI455X (gfx1250) — compile-verified, matrix operand order checked by reading, not run on hardware
//
#include <hip/hip_runtime.h>

typedef _Float16 half_t;
typedef __attribute__((ext_vector_type(16))) _Float16 v16h;
typedef __attribute__((ext_vector_type(2)))  _Float16 h2;
typedef __attribute__((ext_vector_type(8)))  float    v8f;
typedef __attribute__((ext_vector_type(4)))  int      v4i;

typedef __attribute__((address_space(1))) v4i gv4i_t;  // global 16B chunk
typedef __attribute__((address_space(3))) v4i lv4i_t;  // LDS 16B chunk

#define WMMA_F16(a, b, c) \
  __builtin_amdgcn_wmma_f32_16x16x32_f16(false, (a), false, (b), (short)0, (c), false, false)

// Load a 16x32 (f16) WMMA fragment from an LDS row-major matrix [rows][ldk].
// Works for both A (row = Mtile*16 + lane%16) and B (row = pixel = lane%16),
// assuming the symmetric striped layout from the CDNA5 ISA tables:
//   element e<8  -> K = kbase + koff + e          (koff = lane<16 ? 0 : 8)
//   element e>=8 -> K = kbase + 16 + koff + (e-8)
__device__ __forceinline__ v16h frag_ld(const half_t* __restrict__ m, int row,
                                        int ldk, int kbase, int lane) {
  const half_t* r = m + row * ldk + kbase + ((lane & 16) ? 8 : 0);
  v16h f;
#pragma unroll
  for (int e = 0; e < 8; e += 2) {
    h2 lo = *(const h2*)(r + e);
    h2 hi = *(const h2*)(r + 16 + e);
    f[e] = lo[0];  f[e + 1] = lo[1];
    f[e + 8] = hi[0]; f[e + 9] = hi[1];
  }
  return f;
}

// Build a B fragment (K x 16 pixels) from an f32 LDS tile laid out
// [channel][row(8)][col(16)], converting to f16 on the fly.
__device__ __forceinline__ v16h frag_b_tile(const float* __restrict__ t,
                                            int wave, int p, int kbase, int lane) {
  const int koff = kbase + ((lane & 16) ? 8 : 0);
  const float* base = t + wave * 16 + p;
  v16h f;
#pragma unroll
  for (int e = 0; e < 8; ++e) {
    f[e]     = (half_t)base[(koff + e) * (8 * 16)];
    f[e + 8] = (half_t)base[(koff + 16 + e) * (8 * 16)];
  }
  return f;
}

// Convert two f32 D fragments (channel tiles 0..15 / 16..31 of a K-step) into
// the f16 B fragment for the next GEMM. D row layout (M = r + (lane<16?0:8))
// matches the B K layout (K = koff + e / 16 + koff + (e-8)) exactly.
__device__ __forceinline__ v16h dd_to_b(const v8f& lo, const v8f& hi) {
  v16h f;
#pragma unroll
  for (int e = 0; e < 8; ++e) {
    f[e] = (half_t)lo[e];
    f[e + 8] = (half_t)hi[e];
  }
  return f;
}

// ---------------------------------------------------------------------------
// Stage 1: o1 = PReLU(BN1(invo_{s=2}(x)))   x:(8,32,256,256) -> o1:(8,64,128,128)
// Fusion: o1[c] = Winit · ( sum_k kk[k] * x[:, pos_k] ), kk from pooled x.
// Block: 256 threads = 8 waves; wave w handles output row (ytile*8+w),
// 16 consecutive output columns (one WMMA N-tile).
// ---------------------------------------------------------------------------
__global__ __launch_bounds__(256)
void invo_stage1(const float* __restrict__ x,
                 const float* __restrict__ w_init,  // (64,32)
                 const float* __restrict__ w_red,   // (32,32)
                 const float* __restrict__ s_g, const float* __restrict__ s_b,
                 const float* __restrict__ s_m, const float* __restrict__ s_v,
                 const float* __restrict__ w_span,  // (9,32)
                 const float* __restrict__ bn_g, const float* __restrict__ bn_b,
                 const float* __restrict__ bn_m, const float* __restrict__ bn_v,
                 const float* __restrict__ pr,
                 float* __restrict__ o1) {
  __shared__ __align__(16) half_t sWinit[64 * 32];
  __shared__ __align__(16) half_t sWred[32 * 32];
  __shared__ __align__(16) half_t sWspan[16 * 32];     // rows 9..15 zero-padded
  __shared__ float sSsc[32], sSsh[32];                 // inner BN (eps 1e-5)
  __shared__ float sBsc[64], sBsh[64], sBa[64];        // outer BN+PReLU (eps 1e-3)
  __shared__ __align__(16) half_t sAct[8][16][32];     // per-wave [pixel][K]
  __shared__ float sKK[8][16][16];                     // per-wave per-pixel kernels

  const int tid = threadIdx.x;
  const int wave = tid >> 5, lane = tid & 31;
  const int b = blockIdx.z;
  const int ho = blockIdx.y * 8 + wave;
  const int wo0 = blockIdx.x * 16;

  for (int i = tid; i < 64 * 32; i += 256) sWinit[i] = (half_t)w_init[i];
  for (int i = tid; i < 32 * 32; i += 256) sWred[i] = (half_t)w_red[i];
  for (int i = tid; i < 16 * 32; i += 256) {
    int r = i >> 5, c = i & 31;
    sWspan[i] = (r < 9) ? (half_t)w_span[r * 32 + c] : (half_t)0.f;
  }
  for (int i = tid; i < 32; i += 256) {
    float sc = s_g[i] * rsqrtf(s_v[i] + 1e-5f);
    sSsc[i] = sc; sSsh[i] = s_b[i] - s_m[i] * sc;
  }
  for (int i = tid; i < 64; i += 256) {
    float sc = bn_g[i] * rsqrtf(bn_v[i] + 1e-3f);
    sBsc[i] = sc; sBsh[i] = bn_b[i] - bn_m[i] * sc; sBa[i] = pr[i];
  }
  __syncthreads();

  const int p = lane & 15;        // pixel within N-tile
  const int cgrp = lane >> 4;     // channel half handled by this lane
  const int wo = wo0 + p;
  const int rb = (lane & 16) ? 8 : 0;
  const int mrow = lane & 15;

  // ---- Phase O: 2x2 avg-pooled x (32ch) for this wave's 16 pixels -> LDS f16
  {
    const int hi2 = 2 * ho, wi2 = 2 * wo;
    const float* xb = x + ((b * 32 + cgrp * 16) * 256 + hi2) * 256 + wi2;
#pragma unroll 4
    for (int j = 0; j < 16; ++j) {
      const float* pc = xb + j * (256 * 256);
      float2 r0 = *(const float2*)pc;
      float2 r1 = *(const float2*)(pc + 256);
      sAct[wave][p][cgrp * 16 + j] = (half_t)((r0.x + r0.y + r1.x + r1.y) * 0.25f);
    }
  }
  __syncthreads();

  v8f z = {0.f, 0.f, 0.f, 0.f, 0.f, 0.f, 0.f, 0.f};

  // ---- G1: k1 = relu(bn_s(Wred(32x32) * o))   2 M-tiles, K=32
  v16h bO = frag_ld(&sAct[wave][0][0], p, 32, 0, lane);
  v8f d0 = WMMA_F16(frag_ld(sWred, mrow, 32, 0, lane), bO, z);
  v8f d1 = WMMA_F16(frag_ld(sWred, 16 + mrow, 32, 0, lane), bO, z);
#pragma unroll
  for (int r = 0; r < 8; ++r) {
    int c0 = r + rb, c1 = 16 + r + rb;
    d0[r] = fmaxf(d0[r] * sSsc[c0] + sSsh[c0], 0.f);
    d1[r] = fmaxf(d1[r] * sSsc[c1] + sSsh[c1], 0.f);
  }
  // ---- G2: kk = Wspan(9x32, padded to 16) * k1
  v16h bK1 = dd_to_b(d0, d1);
  v8f dk = WMMA_F16(frag_ld(sWspan, mrow, 32, 0, lane), bK1, z);
#pragma unroll
  for (int r = 0; r < 8; ++r) {
    int k = r + rb;
    if (k < 9) sKK[wave][p][k] = dk[r];
  }
  __syncthreads();

  // ---- Phase S: s[ci] = sum_k kk[k] * x[ci, 2ho-1+kh, 2wo-1+kw]  (pad 1)
  float kk[9];
#pragma unroll
  for (int k = 0; k < 9; ++k) kk[k] = sKK[wave][p][k];
  const int r0 = 2 * ho - 1, c0 = 2 * wo - 1;
  bool rv[3], cv[3];
#pragma unroll
  for (int t = 0; t < 3; ++t) {
    rv[t] = (unsigned)(r0 + t) < 256u;
    cv[t] = (unsigned)(c0 + t) < 256u;
  }
  {
    const float* xb = x + (b * 32 + cgrp * 16) * (256 * 256);
#pragma unroll 2
    for (int j = 0; j < 16; ++j) {
      const float* pc = xb + j * (256 * 256);
      float s = 0.f;
#pragma unroll
      for (int kh = 0; kh < 3; ++kh)
#pragma unroll
        for (int kw = 0; kw < 3; ++kw) {
          float v = (rv[kh] && cv[kw]) ? pc[(r0 + kh) * 256 + (c0 + kw)] : 0.f;
          s = fmaf(kk[kh * 3 + kw], v, s);
        }
      sAct[wave][p][cgrp * 16 + j] = (half_t)s;
    }
  }
  __syncthreads();

  // ---- G3: o1 = prelu(bn1(Winit(64x32) * s))   4 M-tiles
  v16h bS = frag_ld(&sAct[wave][0][0], p, 32, 0, lane);
  float* op = o1 + ((b * 64) * 128 + ho) * 128 + wo;
#pragma unroll
  for (int t = 0; t < 4; ++t) {
    v8f acc = WMMA_F16(frag_ld(sWinit, t * 16 + mrow, 32, 0, lane), bS, z);
#pragma unroll
    for (int r = 0; r < 8; ++r) {
      int c = t * 16 + rb + r;
      float v = acc[r] * sBsc[c] + sBsh[c];
      v = (v > 0.f) ? v : sBa[c] * v;
      op[c * (128 * 128)] = v;
    }
  }
}

// ---------------------------------------------------------------------------
// Stage 2: for dilation d (blockIdx.z): y = PReLUf(BNf(PReLUd(BNd(invo_d(o1)))))
// written to out channels [64*z, 64*z+64). o1 stays hot in the 192MB L2.
// The block's o1 tile (64ch x 8row x 16col f32 = 32KB) is staged into LDS with
// async global->LDS copies (no VGPR round-trip, overlaps weight staging).
// ---------------------------------------------------------------------------
__global__ __launch_bounds__(256)
void invo_stage2(const float* __restrict__ o1,
                 const float* __restrict__ wd_red,   // (5,64,64)
                 const float* __restrict__ sd_g, const float* __restrict__ sd_b,
                 const float* __restrict__ sd_m, const float* __restrict__ sd_v,
                 const float* __restrict__ wd_span,  // (5,9,64)
                 const float* __restrict__ bnd_g, const float* __restrict__ bnd_b,
                 const float* __restrict__ bnd_m, const float* __restrict__ bnd_v,
                 const float* __restrict__ prd,
                 const float* __restrict__ bnf_g, const float* __restrict__ bnf_b,
                 const float* __restrict__ bnf_m, const float* __restrict__ bnf_v,
                 const float* __restrict__ prf,
                 float* __restrict__ out) {
  __shared__ __align__(16) half_t sWred[64 * 64];
  __shared__ __align__(16) half_t sWspan[16 * 64];     // rows 9..15 zero-padded
  __shared__ float sSsc[64], sSsh[64];
  __shared__ float sDsc[64], sDsh[64], sDa[64];
  __shared__ float sFsc[64], sFsh[64], sFa[64];
  __shared__ __align__(16) float sTile[64][8][16];     // o1 tile, channel-major f32
  __shared__ float sKK[8][16][16];

  const int tid = threadIdx.x;
  const int wave = tid >> 5, lane = tid & 31;
  const int di = blockIdx.z;
  const int dil = 1 << di;                             // DILATIONS = 1,2,4,8,16
  const int b = blockIdx.y;
  const int xt = blockIdx.x & 7, yt = blockIdx.x >> 3;
  const int ho0 = yt * 8;
  const int ho = ho0 + wave;
  const int wo0 = xt * 16;

  // ---- Async copy of the block's o1 tile into LDS (b128 per item).
  // items: 64ch x 8rows x 4 16-byte chunks = 2048 = 256 threads x 8
  {
    const float* ob = o1 + ((b * 64) * 128 + ho0) * 128 + wo0;
#pragma unroll
    for (int it = 0; it < 8; ++it) {
      int item = tid + it * 256;
      int chunk = item & 3;
      int row = (item >> 2) & 7;
      int ci = item >> 5;
      const float* gp = ob + ci * (128 * 128) + row * 128 + chunk * 4;
      __builtin_amdgcn_global_load_async_to_lds_b128(
          (gv4i_t*)(const_cast<float*>(gp)),
          (lv4i_t*)&sTile[ci][row][chunk * 4], 0, 0);
    }
  }

  for (int i = tid; i < 64 * 64; i += 256) sWred[i] = (half_t)wd_red[di * 64 * 64 + i];
  for (int i = tid; i < 16 * 64; i += 256) {
    int r = i >> 6, c = i & 63;
    sWspan[i] = (r < 9) ? (half_t)wd_span[(di * 9 + r) * 64 + c] : (half_t)0.f;
  }
  for (int i = tid; i < 64; i += 256) {
    int g = di * 64 + i;
    float sc = sd_g[g] * rsqrtf(sd_v[g] + 1e-5f);
    sSsc[i] = sc; sSsh[i] = sd_b[g] - sd_m[g] * sc;
    float dc = bnd_g[g] * rsqrtf(bnd_v[g] + 1e-3f);
    sDsc[i] = dc; sDsh[i] = bnd_b[g] - bnd_m[g] * dc; sDa[i] = prd[g];
    float fc = bnf_g[g] * rsqrtf(bnf_v[g] + 1e-3f);
    sFsc[i] = fc; sFsh[i] = bnf_b[g] - bnf_m[g] * fc; sFa[i] = prf[g];
  }
  __builtin_amdgcn_s_wait_asynccnt(0);
  __syncthreads();

  const int p = lane & 15;
  const int chalf = (lane >> 4) * 32;
  const int rb = (lane & 16) ? 8 : 0;
  const int mrow = lane & 15;
  const int wo = wo0 + p;

  v8f z = {0.f, 0.f, 0.f, 0.f, 0.f, 0.f, 0.f, 0.f};

  // ---- G1: k1 = relu(bn_s(Wred(64x64) * o1))   4 M-tiles x 2 K-steps
  v16h b0 = frag_b_tile(&sTile[0][0][0], wave, p, 0, lane);
  v16h b1 = frag_b_tile(&sTile[0][0][0], wave, p, 32, lane);
  v8f acc[4];
#pragma unroll
  for (int t = 0; t < 4; ++t) {
    acc[t] = WMMA_F16(frag_ld(sWred, t * 16 + mrow, 64, 0, lane), b0, z);
    acc[t] = WMMA_F16(frag_ld(sWred, t * 16 + mrow, 64, 32, lane), b1, acc[t]);
#pragma unroll
    for (int r = 0; r < 8; ++r) {
      int c = t * 16 + rb + r;
      acc[t][r] = fmaxf(acc[t][r] * sSsc[c] + sSsh[c], 0.f);
    }
  }
  // ---- G2: kk = Wspan(9x64) * k1   (2 K-steps)
  v16h bk0 = dd_to_b(acc[0], acc[1]);
  v16h bk1 = dd_to_b(acc[2], acc[3]);
  v8f dk = WMMA_F16(frag_ld(sWspan, mrow, 64, 0, lane), bk0, z);
  dk = WMMA_F16(frag_ld(sWspan, mrow, 64, 32, lane), bk1, dk);
#pragma unroll
  for (int r = 0; r < 8; ++r) {
    int k = r + rb;
    if (k < 9) sKK[wave][p][k] = dk[r];
  }
  __syncthreads();

  // ---- Apply + fused BNd/PReLUd + BNf/PReLUf
  float kk[9];
#pragma unroll
  for (int k = 0; k < 9; ++k) kk[k] = sKK[wave][p][k];
  int hq[3], cq[3];
  bool rv[3], cv[3];
#pragma unroll
  for (int t = 0; t < 3; ++t) {
    hq[t] = ho + (t - 1) * dil; rv[t] = (unsigned)hq[t] < 128u;
    cq[t] = wo + (t - 1) * dil; cv[t] = (unsigned)cq[t] < 128u;
  }
  const float* ob = o1 + (b * 64) * (128 * 128);
  float* outp = out + ((b * 320 + di * 64) * 128 + ho) * 128 + wo;
#pragma unroll 2
  for (int j = 0; j < 32; ++j) {
    int ci = chalf + j;
    const float* pc = ob + ci * (128 * 128);
    float y = 0.f;
#pragma unroll
    for (int kh = 0; kh < 3; ++kh)
#pragma unroll
      for (int kw = 0; kw < 3; ++kw) {
        float v = (rv[kh] && cv[kw]) ? pc[hq[kh] * 128 + cq[kw]] : 0.f;
        y = fmaf(kk[kh * 3 + kw], v, y);
      }
    float v = y * sDsc[ci] + sDsh[ci];
    v = (v > 0.f) ? v : sDa[ci] * v;
    v = v * sFsc[ci] + sFsh[ci];
    v = (v > 0.f) ? v : sFa[ci] * v;
    outp[ci * (128 * 128)] = v;
  }
}

extern "C" void kernel_launch(void* const* d_in, const int* in_sizes, int n_in,
                              void* d_out, int out_size, void* d_ws, size_t ws_size,
                              hipStream_t stream) {
  (void)in_sizes; (void)n_in; (void)out_size; (void)ws_size;
  const float* x       = (const float*)d_in[0];
  const float* w1_init = (const float*)d_in[1];
  const float* w1_red  = (const float*)d_in[2];
  const float* s1_g    = (const float*)d_in[3];
  const float* s1_b    = (const float*)d_in[4];
  const float* s1_m    = (const float*)d_in[5];
  const float* s1_v    = (const float*)d_in[6];
  const float* w1_span = (const float*)d_in[7];
  const float* bn1_g   = (const float*)d_in[8];
  const float* bn1_b   = (const float*)d_in[9];
  const float* bn1_m   = (const float*)d_in[10];
  const float* bn1_v   = (const float*)d_in[11];
  const float* pr1     = (const float*)d_in[12];
  const float* wd_red  = (const float*)d_in[13];
  const float* sd_g    = (const float*)d_in[14];
  const float* sd_b    = (const float*)d_in[15];
  const float* sd_m    = (const float*)d_in[16];
  const float* sd_v    = (const float*)d_in[17];
  const float* wd_span = (const float*)d_in[18];
  const float* bnd_g   = (const float*)d_in[19];
  const float* bnd_b   = (const float*)d_in[20];
  const float* bnd_m   = (const float*)d_in[21];
  const float* bnd_v   = (const float*)d_in[22];
  const float* prd     = (const float*)d_in[23];
  const float* bnf_g   = (const float*)d_in[24];
  const float* bnf_b   = (const float*)d_in[25];
  const float* bnf_m   = (const float*)d_in[26];
  const float* bnf_v   = (const float*)d_in[27];
  const float* prf     = (const float*)d_in[28];

  float* o1ws = (float*)d_ws;          // 8*64*128*128 f32 = 33.5 MB
  float* outp = (float*)d_out;         // (8,320,128,128) f32

  dim3 blk(256);
  // stage1: grid (xtiles=8, ytiles=16, batch=8)
  invo_stage1<<<dim3(8, 16, 8), blk, 0, stream>>>(
      x, w1_init, w1_red, s1_g, s1_b, s1_m, s1_v, w1_span,
      bn1_g, bn1_b, bn1_m, bn1_v, pr1, o1ws);
  // stage2: grid (tiles=128, batch=8, dilation=5)
  invo_stage2<<<dim3(128, 8, 5), blk, 0, stream>>>(
      o1ws, wd_red, sd_g, sd_b, sd_m, sd_v, wd_span,
      bnd_g, bnd_b, bnd_m, bnd_v, prd,
      bnf_g, bnf_b, bnf_m, bnf_v, prf, outp);
}
